// CausalSelfAttention_63642825392583
// MI455X (gfx1250) — matrix operand order repaired and checked by reading, not compiled, not run on hardware
//
#include <hip/hip_runtime.h>

// ---------------------------------------------------------------------------
// CDNA5 (gfx1250) causal self-attention:
//   bf16 pre-convert -> TDM-fed WMMA QKV GEMM -> TDM-fed flash attention
//   -> TDM-fed WMMA out GEMM
// Matmuls: v_wmma_f32_16x16x32_bf16 (wave32). Tiles: tensor_load_to_lds (TDM)
// double-buffered with s_wait_tensorcnt, LDS row padding done by the TDM.
// Fixed problem size: B=2, T=2048, C=1024, H=16, dk=64.
// ---------------------------------------------------------------------------

typedef __attribute__((ext_vector_type(16))) __bf16 v16bf;
typedef __attribute__((ext_vector_type(8)))  float  v8f;
typedef __attribute__((ext_vector_type(4)))  unsigned u32x4;
typedef __attribute__((ext_vector_type(8)))  int      i32x8;
typedef __attribute__((ext_vector_type(4)))  int      i32x4;

union ABFrag { unsigned u[8]; v16bf v; };

#define T_SEQ   2048
#define C_DIM   1024
#define H_HEADS 16
#define DK      64
#define N_QKV   3072
#define LDSTR   36      // GEMM LDS row stride (32 data + 4 pad bf16)
#define KSTR    66      // attn K-tile LDS row stride (64 + 2)
#define VSTR    34      // attn V-tile LDS row stride (32 + 2)

#define HAVE_TDM __has_builtin(__builtin_amdgcn_tensor_load_to_lds)

__device__ __forceinline__ unsigned short f2bf_bits(float f) {
  __bf16 h = (__bf16)f;
  return __builtin_bit_cast(unsigned short, h);
}
__device__ __forceinline__ unsigned pack_bf16x2(float lo, float hi) {
  return (unsigned)f2bf_bits(lo) | ((unsigned)f2bf_bits(hi) << 16);
}
__device__ __forceinline__ v8f wmma_bf16(const ABFrag& a, const ABFrag& b, v8f c) {
  return __builtin_amdgcn_wmma_f32_16x16x32_bf16(false, a.v, false, b.v,
                                                 (short)0, c, false, false);
}
__device__ __forceinline__ unsigned lds_addr_of(const void* p) {
  return (unsigned)(unsigned long long)(size_t)p;   // low 32 bits = LDS offset
}

#if HAVE_TDM
// 2D bf16 tile load, global -> LDS, via Tensor Data Mover.
// tile_w elems contiguous (dim0), tile_h rows, row pitch row_stride elems.
// pad_interval/pad_amount: LDS padding per D# group1 encoding
// (pad after 2<<interval DWORDs, pad of (amount+1) DWORDs).
__device__ __forceinline__ void tdm_tile_load(const void* gaddr, unsigned ldsoff,
                                              unsigned tile_w, unsigned tile_h,
                                              unsigned row_stride_elems,
                                              int pad_interval, int pad_amount) {
  unsigned long long ga = (unsigned long long)(size_t)gaddr;
  u32x4 g0;
  g0[0] = 1u;                                   // count=1: valid user descriptor
  g0[1] = ldsoff;                               // lds_addr (bytes)
  g0[2] = (unsigned)ga;                         // global_addr[31:0]
  g0[3] = (unsigned)(ga >> 32) | (2u << 30);    // global_addr[56:32] | type=2
  i32x8 g1;
  g1[0] = (1 << 16)                             // data_size = 2 bytes
        | (1 << 20)                             // pad_enable
        | (pad_interval << 22) | (pad_amount << 25);
  g1[1] = (int)((tile_w & 0xFFFFu) << 16);                    // tensor_dim0 lo
  g1[2] = (int)((tile_w >> 16) | ((tile_h & 0xFFFFu) << 16)); // dim0 hi|dim1 lo
  g1[3] = (int)((tile_h >> 16) | (tile_w << 16));             // dim1 hi|tile_dim0
  g1[4] = (int)(tile_h & 0xFFFFu);                            // tile_dim1, dim2=0
  g1[5] = (int)row_stride_elems;                              // dim0_stride lo
  g1[6] = 0;
  g1[7] = 0;
  i32x4 z4 = {0, 0, 0, 0};
#if __clang_major__ >= 23
  i32x8 z8 = {0, 0, 0, 0, 0, 0, 0, 0};
  __builtin_amdgcn_tensor_load_to_lds(g0, g1, z4, z4, z8, 0);
#else
  __builtin_amdgcn_tensor_load_to_lds(g0, g1, z4, z4, 0);
#endif
}
#endif

// Fragment loads (pairs of bf16 as u32, element s at k = kb + s, kb = 16*hi: contiguous k order in A and B)
__device__ __forceinline__ void load_a_frag(const __bf16* src, int kb, ABFrag& f) {
#pragma unroll
  for (int j = 0; j < 4; ++j) {
    f.u[j]     = *(const unsigned*)(src + kb + 2 * j);
    f.u[j + 4] = *(const unsigned*)(src + 8 + kb + 2 * j);
  }
}
__device__ __forceinline__ void load_b_frag(const __bf16* src, int kb, ABFrag& f) {
#pragma unroll
  for (int j = 0; j < 8; ++j)
    f.u[j] = *(const unsigned*)(src + kb + 2 * j);
}

// ---------------------------------------------------------------------------
// Shared GEMM pipeline: acc(128x128) += A(128x1024) * B^T(128x1024)
// A rows: Ag + row*1024; B rows (pre-transposed weights): Bg + row*1024.
// TDM double-buffered; 32 K-steps of 32.
// ---------------------------------------------------------------------------
__device__ __forceinline__ void gemm_core(
    const __bf16* __restrict__ Ag, const __bf16* __restrict__ Bg,
    __bf16 (*SA)[128 * LDSTR], __bf16 (*SB)[128 * LDSTR],
    int tid, int wave, int l15, int hi, int waveM, int waveN, v8f acc[4][2])
{
  const int NSTEP = C_DIM / 32;
#if HAVE_TDM
  if (wave == 0) {
    tdm_tile_load(Ag, lds_addr_of(&SA[0][0]), 32, 128, C_DIM, 3, 1);
    tdm_tile_load(Bg, lds_addr_of(&SB[0][0]), 32, 128, C_DIM, 3, 1);
  }
#endif
  for (int ks = 0; ks < NSTEP; ++ks) {
    int buf;
#if HAVE_TDM
    buf = ks & 1;
    if (wave == 0) {
      if (ks + 1 < NSTEP) {
        tdm_tile_load(Ag + (ks + 1) * 32, lds_addr_of(&SA[buf ^ 1][0]), 32, 128, C_DIM, 3, 1);
        tdm_tile_load(Bg + (ks + 1) * 32, lds_addr_of(&SB[buf ^ 1][0]), 32, 128, C_DIM, 3, 1);
        __builtin_amdgcn_s_wait_tensorcnt(2);   // current step's two tiles done
      } else {
        __builtin_amdgcn_s_wait_tensorcnt(0);
      }
    }
    __syncthreads();                            // release consumers
#else
    buf = 0;
    int k0 = ks * 32;
    __syncthreads();
#pragma unroll
    for (int it = 0; it < 4; ++it) {
      int idx = tid + it * 256;
      int row = idx >> 3;
      int kk  = (idx & 7) * 4;
      *(uint2*)(&SA[0][row * LDSTR + kk]) = *(const uint2*)(Ag + (size_t)row * C_DIM + k0 + kk);
      *(uint2*)(&SB[0][row * LDSTR + kk]) = *(const uint2*)(Bg + (size_t)row * C_DIM + k0 + kk);
    }
    __syncthreads();
#endif
    ABFrag a[4], b[2];
#pragma unroll
    for (int wm = 0; wm < 4; ++wm)
      load_a_frag(&SA[buf][(waveM * 64 + wm * 16 + l15) * LDSTR], hi * 16, a[wm]);
#pragma unroll
    for (int wn = 0; wn < 2; ++wn)
      load_b_frag(&SB[buf][(waveN * 32 + wn * 16 + l15) * LDSTR], hi * 16, b[wn]);
#pragma unroll
    for (int wm = 0; wm < 4; ++wm)
#pragma unroll
      for (int wn = 0; wn < 2; ++wn)
        acc[wm][wn] = wmma_bf16(a[wm], b[wn], acc[wm][wn]);
#if HAVE_TDM
    __syncthreads();                            // all readers done before reuse
#endif
  }
}

// ---------------------------------------------------------------------------
// Pre-conversion kernels
// ---------------------------------------------------------------------------
__global__ __launch_bounds__(256, 1) void cvt_kernel(
    const float* __restrict__ in, __bf16* __restrict__ out, int n4) {
  int i = blockIdx.x * 256 + threadIdx.x;
  if (i < n4) {
    float4 f = *(const float4*)(in + (size_t)i * 4);
    uint2 p;
    p.x = pack_bf16x2(f.x, f.y);
    p.y = pack_bf16x2(f.z, f.w);
    *(uint2*)(out + (size_t)i * 4) = p;
  }
}

// in: K x N f32 row-major -> out: N x K bf16 (transposed)
__global__ __launch_bounds__(256, 1) void transpose_cvt_kernel(
    const float* __restrict__ in, __bf16* __restrict__ out, int K, int N) {
  long long idx = (long long)blockIdx.x * 256 + threadIdx.x;
  int k = (int)(idx / N);
  int n = (int)(idx % N);
  out[(size_t)n * K + k] = (__bf16)in[idx];
}

// ---------------------------------------------------------------------------
// Kernel 1: qkv = xb @ wqkvT^T + b_qkv, scattered to Q/K (bh,t,d), V^T (bh,d,t)
// ---------------------------------------------------------------------------
__global__ __launch_bounds__(256, 1) void qkv_gemm_kernel(
    const __bf16* __restrict__ xb, const __bf16* __restrict__ wT,
    const float* __restrict__ bias,
    __bf16* __restrict__ qws, __bf16* __restrict__ kws, __bf16* __restrict__ vtws)
{
  __shared__ __align__(16) __bf16 SA[2][128 * LDSTR];
  __shared__ __align__(16) __bf16 SB[2][128 * LDSTR];

  const int tid  = threadIdx.x;
  const int wave = tid >> 5, lane = tid & 31;
  const int l15  = lane & 15, hi = lane >> 4;
  const int waveM = wave >> 2, waveN = wave & 3;
  const int blockM = blockIdx.y * 128, blockN = blockIdx.x * 128;

  v8f acc[4][2];
  v8f zf = {};
#pragma unroll
  for (int i = 0; i < 4; ++i)
#pragma unroll
    for (int j = 0; j < 2; ++j) acc[i][j] = zf;

  gemm_core(xb + (size_t)blockM * C_DIM, wT + (size_t)blockN * C_DIM,
            SA, SB, tid, wave, l15, hi, waveM, waveN, acc);

#pragma unroll
  for (int wm = 0; wm < 4; ++wm)
#pragma unroll
    for (int wn = 0; wn < 2; ++wn) {
      int col = blockN + waveN * 32 + wn * 16 + l15;     // 0..3071
      int h   = col / 192;
      int j   = col % 192;
      int sel = j >> 6;                                   // 0=Q 1=K 2=V
      int d   = j & 63;
      float bb = bias[col];
#pragma unroll
      for (int r = 0; r < 8; ++r) {
        int row = blockM + waveM * 64 + wm * 16 + r + hi * 8;
        int bi  = row >> 11;
        int t   = row & (T_SEQ - 1);
        int bh  = bi * H_HEADS + h;
        __bf16 v = (__bf16)(acc[wm][wn][r] + bb);
        if (sel == 0)      qws [((size_t)bh * T_SEQ + t) * DK + d] = v;
        else if (sel == 1) kws [((size_t)bh * T_SEQ + t) * DK + d] = v;
        else               vtws[((size_t)bh * DK + d) * T_SEQ + t] = v;
      }
    }
}

// ---------------------------------------------------------------------------
// Kernel 2: flash attention. Block = 8 waves sharing one bh; wave w owns
// query tile qt = blk*8+w (16 rows). K/V tiles staged via TDM into LDS and
// reused by all waves; uniform barrier schedule to the block's max range.
// ---------------------------------------------------------------------------
__global__ __launch_bounds__(256, 1) void attn_kernel(
    const __bf16* __restrict__ qws, const __bf16* __restrict__ kws,
    const __bf16* __restrict__ vtws, __bf16* __restrict__ attnws)
{
  __shared__ __align__(16) __bf16 Kt[2][32 * KSTR];   // [key][d]
  __shared__ __align__(16) __bf16 Vt[2][64 * VSTR];   // [d][key]
  __shared__ __align__(8)  __bf16 Pl[8][16 * LDSTR];  // per-wave P scratch

  const int tid  = threadIdx.x;
  const int wave = tid >> 5, lane = tid & 31;
  const int l15  = lane & 15, hi = lane >> 4;
  const int bh   = blockIdx.x >> 4;
  const int blk  = blockIdx.x & 15;
  const int qt   = blk * 8 + wave;
  const int q0   = qt * 16;
  const int kend = q0 + 16;
  const int nsteps = blk * 4 + 4;                     // (blk*128+128)/32

  const __bf16* kg = kws  + (size_t)bh * T_SEQ * DK;  // (t, d)
  const __bf16* vg = vtws + (size_t)bh * DK * T_SEQ;  // (d, t)

  // Resident Q fragments (16 rows x 64 d = two A frags)
  ABFrag qf[2];
  {
    const __bf16* qp = qws + ((size_t)bh * T_SEQ + q0 + l15) * DK;
    load_a_frag(qp,      hi * 16, qf[0]);
    load_a_frag(qp + 32, hi * 16, qf[1]);
  }

  v8f o[4];
  v8f zf = {};
#pragma unroll
  for (int g = 0; g < 4; ++g) o[g] = zf;
  float m[8], ls[8];
#pragma unroll
  for (int r = 0; r < 8; ++r) { m[r] = -__builtin_inff(); ls[r] = 0.0f; }

#if HAVE_TDM
  if (wave == 0) {
    tdm_tile_load(kg, lds_addr_of(&Kt[0][0]), 64, 32, DK,    4, 0);
    tdm_tile_load(vg, lds_addr_of(&Vt[0][0]), 32, 64, T_SEQ, 3, 0);
  }
#endif
  for (int ks = 0; ks < nsteps; ++ks) {
    const int kb0 = ks * 32;
    int buf;
#if HAVE_TDM
    buf = ks & 1;
    if (wave == 0) {
      if (ks + 1 < nsteps) {
        tdm_tile_load(kg + (size_t)(kb0 + 32) * DK, lds_addr_of(&Kt[buf ^ 1][0]), 64, 32, DK,    4, 0);
        tdm_tile_load(vg + (kb0 + 32),              lds_addr_of(&Vt[buf ^ 1][0]), 32, 64, T_SEQ, 3, 0);
        __builtin_amdgcn_s_wait_tensorcnt(2);
      } else {
        __builtin_amdgcn_s_wait_tensorcnt(0);
      }
    }
    __syncthreads();
#else
    buf = 0;
    __syncthreads();
#pragma unroll
    for (int it = 0; it < 2; ++it) {
      int idx = tid + it * 256;                 // 0..511
      {
        int r = idx >> 4, d4 = (idx & 15) * 4;  // K tile 32x64
        uint2 v = *(const uint2*)(kg + (size_t)(kb0 + r) * DK + d4);
        *(unsigned*)(&Kt[0][r * KSTR + d4])     = v.x;
        *(unsigned*)(&Kt[0][r * KSTR + d4 + 2]) = v.y;
      }
      {
        int r = idx >> 3, k4 = (idx & 7) * 4;   // V tile 64x32
        uint2 v = *(const uint2*)(vg + (size_t)r * T_SEQ + kb0 + k4);
        *(unsigned*)(&Vt[0][r * VSTR + k4])     = v.x;
        *(unsigned*)(&Vt[0][r * VSTR + k4 + 2]) = v.y;
      }
    }
    __syncthreads();
#endif

    if (kb0 < kend) {
      // ---- S = Q K^T (16 x 32): two C frags, chained over dk
      v8f s[2];
#pragma unroll
      for (int sf = 0; sf < 2; ++sf) {
        const __bf16* kp = &Kt[buf][(sf * 16 + l15) * KSTR];
        ABFrag b0, b1;
        load_b_frag(kp,      hi * 16, b0);      // d in [0,32)
        load_b_frag(kp + 32, hi * 16, b1);      // d in [32,64)
        v8f z = {};
        s[sf] = wmma_bf16(qf[0], b0, z);
        s[sf] = wmma_bf16(qf[1], b1, s[sf]);
      }

      // ---- scale, causal mask, tile row-max (16-lane xor reduce)
      float tmax[8];
#pragma unroll
      for (int r = 0; r < 8; ++r) {
        int qrow = q0 + r + hi * 8;
        float s0 = s[0][r] * 0.125f;
        float s1 = s[1][r] * 0.125f;
        if (kb0 + l15 > qrow)      s0 = -__builtin_inff();
        if (kb0 + 16 + l15 > qrow) s1 = -__builtin_inff();
        s[0][r] = s0; s[1][r] = s1;
        float mx = fmaxf(s0, s1);
        mx = fmaxf(mx, __shfl_xor(mx, 1, 32));
        mx = fmaxf(mx, __shfl_xor(mx, 2, 32));
        mx = fmaxf(mx, __shfl_xor(mx, 4, 32));
        mx = fmaxf(mx, __shfl_xor(mx, 8, 32));
        tmax[r] = mx;
      }

      // ---- online softmax update
      float alpha[8];
#pragma unroll
      for (int r = 0; r < 8; ++r) {
        float mn = fmaxf(m[r], tmax[r]);
        alpha[r] = __expf(m[r] - mn);
        float p0 = __expf(s[0][r] - mn);
        float p1 = __expf(s[1][r] - mn);
        s[0][r] = p0; s[1][r] = p1;
        float sm = p0 + p1;
        sm += __shfl_xor(sm, 1, 32);
        sm += __shfl_xor(sm, 2, 32);
        sm += __shfl_xor(sm, 4, 32);
        sm += __shfl_xor(sm, 8, 32);
        ls[r] = ls[r] * alpha[r] + sm;
        m[r]  = mn;
      }
#pragma unroll
      for (int g = 0; g < 4; ++g)
#pragma unroll
        for (int r = 0; r < 8; ++r) o[g][r] *= alpha[r];

      // ---- P: C-layout -> A-layout via per-wave LDS (same-wave DS in order)
      __bf16* pl = &Pl[wave][0];
#pragma unroll
      for (int r = 0; r < 8; ++r) {
        int row = r + hi * 8;
        pl[row * LDSTR + l15]      = (__bf16)s[0][r];
        pl[row * LDSTR + 16 + l15] = (__bf16)s[1][r];
      }
      ABFrag pa;
      load_a_frag(pl + l15 * LDSTR, hi * 16, pa);

      // ---- O += P @ V from LDS V tile
#pragma unroll
      for (int g = 0; g < 4; ++g) {
        ABFrag bv;
        load_b_frag(&Vt[buf][(g * 16 + l15) * VSTR], hi * 16, bv);
        o[g] = wmma_bf16(pa, bv, o[g]);
      }
    }
    __syncthreads();                            // readers done before tile reuse
  }

  // ---- normalize and store to attn workspace (b, t, h*64+d) bf16
  float inv[8];
#pragma unroll
  for (int r = 0; r < 8; ++r) inv[r] = 1.0f / ls[r];
  int b = bh >> 4, h = bh & 15;
#pragma unroll
  for (int g = 0; g < 4; ++g) {
    int c = h * DK + g * 16 + l15;
#pragma unroll
    for (int r = 0; r < 8; ++r) {
      int t = q0 + r + hi * 8;
      attnws[((size_t)(b * T_SEQ + t)) * C_DIM + c] = (__bf16)(o[g][r] * inv[r]);
    }
  }
}

// ---------------------------------------------------------------------------
// Kernel 3: out = attn @ w_out + b_out (f32 output)
// ---------------------------------------------------------------------------
__global__ __launch_bounds__(256, 1) void out_gemm_kernel(
    const __bf16* __restrict__ A, const __bf16* __restrict__ wT,
    const float* __restrict__ bias, float* __restrict__ Out)
{
  __shared__ __align__(16) __bf16 SA[2][128 * LDSTR];
  __shared__ __align__(16) __bf16 SB[2][128 * LDSTR];

  const int tid  = threadIdx.x;
  const int wave = tid >> 5, lane = tid & 31;
  const int l15  = lane & 15, hi = lane >> 4;
  const int waveM = wave >> 2, waveN = wave & 3;
  const int blockM = blockIdx.y * 128, blockN = blockIdx.x * 128;

  v8f acc[4][2];
  v8f zf = {};
#pragma unroll
  for (int i = 0; i < 4; ++i)
#pragma unroll
    for (int j = 0; j < 2; ++j) acc[i][j] = zf;

  gemm_core(A + (size_t)blockM * C_DIM, wT + (size_t)blockN * C_DIM,
            SA, SB, tid, wave, l15, hi, waveM, waveN, acc);

#pragma unroll
  for (int wm = 0; wm < 4; ++wm)
#pragma unroll
    for (int wn = 0; wn < 2; ++wn) {
      int col = blockN + waveN * 32 + wn * 16 + l15;
      float bb = bias[col];
#pragma unroll
      for (int r = 0; r < 8; ++r) {
        int row = blockM + waveM * 64 + wm * 16 + r + hi * 8;
        Out[(size_t)row * C_DIM + col] = acc[wm][wn][r] + bb;
      }
    }
}

// ---------------------------------------------------------------------------
// Host launcher
// ---------------------------------------------------------------------------
extern "C" void kernel_launch(void* const* d_in, const int* in_sizes, int n_in,
                              void* d_out, int out_size, void* d_ws, size_t ws_size,
                              hipStream_t stream) {
  const float* x     = (const float*)d_in[0];   // (2,2048,1024)
  const float* w_qkv = (const float*)d_in[1];   // (1024,3072)
  const float* b_qkv = (const float*)d_in[2];   // (3072,)
  const float* w_out = (const float*)d_in[3];   // (1024,1024)
  const float* b_out = (const float*)d_in[4];   // (1024,)
  float* out = (float*)d_out;                   // (2,2048,1024)

  char* ws = (char*)d_ws;
  __bf16* xb     = (__bf16*)(ws);                        //  8 MB: x bf16
  __bf16* wqkvT  = (__bf16*)(ws + (8u  << 20));          //  6 MB: w_qkv^T bf16
  __bf16* woutT  = (__bf16*)(ws + (14u << 20));          //  2 MB: w_out^T bf16
  __bf16* qws    = (__bf16*)(ws + (16u << 20));          //  8 MB
  __bf16* kws    = (__bf16*)(ws + (24u << 20));          //  8 MB
  __bf16* vtws   = (__bf16*)(ws + (32u << 20));          //  8 MB
  __bf16* attnws = (__bf16*)(ws + (40u << 20));          //  8 MB

  cvt_kernel<<<4096, 256, 0, stream>>>(x, xb, 1048576);             // 4M elems /4
  transpose_cvt_kernel<<<12288, 256, 0, stream>>>(w_qkv, wqkvT, C_DIM, N_QKV);
  transpose_cvt_kernel<<<4096, 256, 0, stream>>>(w_out, woutT, C_DIM, C_DIM);

  qkv_gemm_kernel<<<dim3(N_QKV / 128, 4096 / 128), 256, 0, stream>>>(
      xb, wqkvT, b_qkv, qws, kws, vtws);
  attn_kernel<<<dim3(32 * 16), 256, 0, stream>>>(qws, kws, vtws, attnws);
  out_gemm_kernel<<<dim3(C_DIM / 128, 4096 / 128), 256, 0, stream>>>(
      attnws, woutT, b_out, out);
}
